// trajectory2seq_attn_bi_3058016715236
// MI455X (gfx1250) — compile-verified, matrix operand order checked by reading, not run on hardware
//
#include <hip/hip_runtime.h>
#include <stdint.h>

// ---------------- types ----------------
typedef __attribute__((ext_vector_type(16))) __bf16 v16bf;
typedef __attribute__((ext_vector_type(8)))  float  v8f;
typedef __attribute__((ext_vector_type(4)))  unsigned int v4u;
typedef __attribute__((ext_vector_type(2)))  unsigned int v2u;

#define WMMA_BF16(a, b, c) \
  __builtin_amdgcn_wmma_f32_16x16x32_bf16(false, (a), false, (b), (short)0, (c), false, false)

__device__ __forceinline__ uint16_t f2bf(float f) {
  uint32_t u = __builtin_bit_cast(uint32_t, f);
  uint32_t r = u + 0x7FFFu + ((u >> 16) & 1u);   // round-to-nearest-even
  return (uint16_t)(r >> 16);
}
__device__ __forceinline__ float bf2f(uint16_t h) {
  return __builtin_bit_cast(float, (uint32_t)h << 16);
}

// WMMA 16x16x32 bf16 operand fragment (documented ISA layout):
// lane = (row/col & 15) + 16*khalf; elems 0..7 = K[kb + 8*khalf .. +7],
// elems 8..15 = K[kb + 16 + 8*khalf .. +7]  -> two b128 loads per lane.
__device__ __forceinline__ v16bf load_frag(const uint16_t* p, int lane) {
  const v4u* pp = (const v4u*)(p + 8 * (lane >> 4));
  union { v4u q[2]; v16bf v; } u;
  u.q[0] = pp[0];
  u.q[1] = pp[2];   // +16 elements (= +32 bytes)
  return u.v;
}

// grid-wide generation barrier (all blocks co-resident; bar[0]=cnt, bar[1]=gen)
__device__ __forceinline__ void gbar(int* bar, int nb) {
  __syncthreads();
  __threadfence();
  if (threadIdx.x == 0) {
    int g = atomicAdd(&bar[1], 0);
    if (atomicAdd(&bar[0], 1) == nb - 1) {
      bar[0] = 0;
      __threadfence();
      atomicAdd(&bar[1], 1);
    } else {
      while (atomicAdd(&bar[1], 0) == g) __builtin_amdgcn_s_sleep(1);
    }
  }
  __syncthreads();
}

// ---------------- utility kernels ----------------
__global__ void cvt_bf(const float* __restrict__ s, uint16_t* __restrict__ d, int n) {
  int i = blockIdx.x * blockDim.x + threadIdx.x;
  if (i < n) d[i] = f2bf(s[i]);
}
__global__ void zero32(int* p, int n) {
  int i = blockIdx.x * blockDim.x + threadIdx.x;
  if (i < n) p[i] = 0;
}
__global__ void copyf(const float* __restrict__ s, float* __restrict__ d, int n) {
  int i = blockIdx.x * blockDim.x + threadIdx.x;
  if (i < n) d[i] = s[i];
}
__global__ void r2hdec(const float* __restrict__ R, float* __restrict__ hf,
                       uint16_t* __restrict__ hb, int n) {
  int i = blockIdx.x * blockDim.x + threadIdx.x;
  if (i < n) { float v = R[i]; hf[i] = v; hb[i] = f2bf(v); }
}
__global__ void dec_gather(const uint16_t* __restrict__ embbf, const int* __restrict__ tok,
                           uint16_t* __restrict__ xinbf) {
  int i = blockIdx.x * blockDim.x + threadIdx.x;   // 64*512
  if (i < 64 * 512) {
    int b = i >> 9, k = i & 511;
    xinbf[i] = embbf[tok[b] * 512 + k];
  }
}

// ---------------- generic (dual) WMMA GEMM: C = act(A1*B1^T + A2*B2^T + bias) ----------------
__global__ void __launch_bounds__(256) wmma_gemm_dual(
    const uint16_t* __restrict__ A1, int lda1, const uint16_t* __restrict__ B1, int ldb1, int K1,
    const uint16_t* __restrict__ A2, int lda2, const uint16_t* __restrict__ B2, int ldb2, int K2,
    const float* __restrict__ bias1, const float* __restrict__ bias2,
    float* __restrict__ C, int ldc, uint16_t* __restrict__ Cbf, int ldcb,
    int M, int N, int act)
{
  const int lane = threadIdx.x & 31;
  const int wave = (blockIdx.x * blockDim.x + threadIdx.x) >> 5;
  const int nwaves = (gridDim.x * blockDim.x) >> 5;
  const int mt = M >> 4, nt = N >> 4;
  for (int tile = wave; tile < mt * nt; tile += nwaves) {
    const int m0 = (tile / nt) << 4;
    const int n0 = (tile % nt) << 4;
    v8f acc = {};
    const uint16_t* ar = A1 + (size_t)(m0 + (lane & 15)) * lda1;
    const uint16_t* br = B1 + (size_t)(n0 + (lane & 15)) * ldb1;
    for (int kb = 0; kb < K1; kb += 32) {
      __builtin_prefetch(br + kb + 256, 0, 1);
      v16bf a = load_frag(ar + kb, lane);
      v16bf b = load_frag(br + kb, lane);
      acc = WMMA_BF16(a, b, acc);
    }
    if (A2) {
      const uint16_t* ar2 = A2 + (size_t)(m0 + (lane & 15)) * lda2;
      const uint16_t* br2 = B2 + (size_t)(n0 + (lane & 15)) * ldb2;
      for (int kb = 0; kb < K2; kb += 32) {
        v16bf a = load_frag(ar2 + kb, lane);
        v16bf b = load_frag(br2 + kb, lane);
        acc = WMMA_BF16(a, b, acc);
      }
    }
    const int col = n0 + (lane & 15);
    float bsum = (bias1 ? bias1[col] : 0.f) + (bias2 ? bias2[col] : 0.f);
    for (int r = 0; r < 8; ++r) {
      int row = m0 + r + ((lane >> 4) << 3);
      float v = acc[r] + bsum;
      if (act == 1) v = tanhf(v);
      if (C)   C[(size_t)row * ldc + col] = v;
      if (Cbf) Cbf[(size_t)row * ldcb + col] = f2bf(v);
    }
  }
}

// ---------------- encoder layer 0 (persistent, both dirs; 32 blocks) ----------------
// x:[64,2,1024]  Wih0:[2,1536,2]  Whhbf:[2,1536,512]
// hbf:[2 parity][2 dir][64][512] (pre-zeroed)  ybf:[1024][64][1024]  henc slots 0,1
__global__ void __launch_bounds__(256) enc_gru0(
    const float* __restrict__ x, const float* __restrict__ Wih0,
    const float* __restrict__ bih0, const float* __restrict__ bhh0,
    const uint16_t* __restrict__ Whhbf,
    uint16_t* __restrict__ hbf, uint16_t* __restrict__ ybf,
    float* __restrict__ henc, int* bar)
{
  const int dir = blockIdx.x >> 4;
  const int bid = blockIdx.x & 15;
  const int lane = threadIdx.x & 31;
  const int wv = threadIdx.x >> 5;
  const int hc0 = bid * 32;                      // this block owns h-cols [hc0, hc0+32)
  __shared__ float hloc[64 * 32];
  __shared__ float gh[3][64][32];
  for (int i = threadIdx.x; i < 64 * 32; i += 256) hloc[i] = 0.f;
  __syncthreads();
  const uint16_t* Wd = Whhbf + (size_t)dir * (1536 * 512);
  const float* Wi = Wih0 + dir * 3072;
  const float* bi = bih0 + dir * 1536;
  const float* bh = bhh0 + dir * 1536;

  for (int tt = 0; tt < 1024; ++tt) {
    const int t = dir ? (1023 - tt) : tt;
    const int p = tt & 1;
    const uint16_t* hA = hbf + p * 65536 + dir * 32768;
    uint16_t* hW = hbf + (1 - p) * 65536 + dir * 32768;
    // 24 tiles: 4 m-tiles x 6 n-tiles (gate = nj>>1)
    for (int tile = wv; tile < 24; tile += 8) {
      const int mi = tile / 6, nj = tile % 6;
      const int g = nj >> 1;
      const int n0 = g * 512 + hc0 + ((nj & 1) << 4);
      const int m0 = mi << 4;
      v8f acc = {};
      const uint16_t* ar = hA + (m0 + (lane & 15)) * 512;
      const uint16_t* br = Wd + (size_t)(n0 + (lane & 15)) * 512;
      for (int kb = 0; kb < 512; kb += 32) {
        __builtin_prefetch(br + kb + 256, 0, 1);
        v16bf a = load_frag(ar + kb, lane);
        v16bf b = load_frag(br + kb, lane);
        acc = WMMA_BF16(a, b, acc);
      }
      const int lc = ((nj & 1) << 4) + (lane & 15);
      for (int r = 0; r < 8; ++r)
        gh[g][m0 + r + ((lane >> 4) << 3)][lc] = acc[r];
    }
    __syncthreads();
    // GRU elementwise (input contribution has K=2 -> scalar)
    for (int idx = threadIdx.x; idx < 64 * 32; idx += 256) {
      const int m = idx >> 5, lc = idx & 31;
      const int col = hc0 + lc;
      const float x0 = x[m * 2048 + t];
      const float x1 = x[m * 2048 + 1024 + t];
      const float gir = x0 * Wi[col * 2] + x1 * Wi[col * 2 + 1] + bi[col];
      const float giz = x0 * Wi[(512 + col) * 2] + x1 * Wi[(512 + col) * 2 + 1] + bi[512 + col];
      const float gin = x0 * Wi[(1024 + col) * 2] + x1 * Wi[(1024 + col) * 2 + 1] + bi[1024 + col];
      const float r_ = 1.f / (1.f + __expf(-(gir + gh[0][m][lc] + bh[col])));
      const float z_ = 1.f / (1.f + __expf(-(giz + gh[1][m][lc] + bh[512 + col])));
      const float n_ = tanhf(gin + r_ * (gh[2][m][lc] + bh[1024 + col]));
      const float hn = (1.f - z_) * n_ + z_ * hloc[idx];
      hloc[idx] = hn;
      const uint16_t hb16 = f2bf(hn);
      hW[m * 512 + col] = hb16;
      ybf[(size_t)t * 65536 + m * 1024 + dir * 512 + col] = hb16;
    }
    gbar(bar, gridDim.x);
  }
  for (int idx = threadIdx.x; idx < 64 * 32; idx += 256) {
    const int m = idx >> 5, lc = idx & 31;
    henc[dir * 32768 + m * 512 + hc0 + lc] = hloc[idx];
  }
}

// ---------------- encoder layer 1 (persistent; input GEMM K=1024 + recurrent K=512) ----------------
// n-gate needs gi and gh kept separate -> 4 LDS planes.
__global__ void __launch_bounds__(256) enc_gru1(
    const uint16_t* __restrict__ y0, const uint16_t* __restrict__ Wihbf,
    const float* __restrict__ bih, const float* __restrict__ bhh,
    const uint16_t* __restrict__ Whhbf,
    uint16_t* __restrict__ hbf, uint16_t* __restrict__ y1,
    float* __restrict__ henc, int* bar)
{
  const int dir = blockIdx.x >> 4;
  const int bid = blockIdx.x & 15;
  const int lane = threadIdx.x & 31;
  const int wv = threadIdx.x >> 5;
  const int hc0 = bid * 32;
  __shared__ float hloc[64 * 32];
  __shared__ float gh[4][64][32];
  for (int i = threadIdx.x; i < 64 * 32; i += 256) hloc[i] = 0.f;
  __syncthreads();
  const uint16_t* Wd = Whhbf + (size_t)dir * (1536 * 512);
  const uint16_t* Wi = Wihbf + (size_t)dir * (1536 * 1024);
  const float* bi = bih + dir * 1536;
  const float* bh = bhh + dir * 1536;

  for (int tt = 0; tt < 1024; ++tt) {
    const int t = dir ? (1023 - tt) : tt;
    const int p = tt & 1;
    const uint16_t* hA = hbf + p * 65536 + dir * 32768;
    uint16_t* hW = hbf + (1 - p) * 65536 + dir * 32768;
    const uint16_t* yA = y0 + (size_t)t * 65536;
    for (int tile = wv; tile < 24; tile += 8) {
      const int mi = tile / 6, nj = tile % 6;
      const int g = nj >> 1;
      const int n0 = g * 512 + hc0 + ((nj & 1) << 4);
      const int m0 = mi << 4;
      const int lc = ((nj & 1) << 4) + (lane & 15);
      const uint16_t* ar = yA + (m0 + (lane & 15)) * 1024;
      const uint16_t* br = Wi + (size_t)(n0 + (lane & 15)) * 1024;
      const uint16_t* ar2 = hA + (m0 + (lane & 15)) * 512;
      const uint16_t* br2 = Wd + (size_t)(n0 + (lane & 15)) * 512;
      if (g < 2) {            // r,z gates: single fused accumulator
        v8f acc = {};
        for (int kb = 0; kb < 1024; kb += 32) {
          __builtin_prefetch(br + kb + 256, 0, 1);
          acc = WMMA_BF16(load_frag(ar + kb, lane), load_frag(br + kb, lane), acc);
        }
        for (int kb = 0; kb < 512; kb += 32)
          acc = WMMA_BF16(load_frag(ar2 + kb, lane), load_frag(br2 + kb, lane), acc);
        for (int r = 0; r < 8; ++r)
          gh[g][m0 + r + ((lane >> 4) << 3)][lc] = acc[r];
      } else {                // n gate: keep input/recurrent separate
        v8f ai = {}; v8f ah = {};
        for (int kb = 0; kb < 1024; kb += 32) {
          __builtin_prefetch(br + kb + 256, 0, 1);
          ai = WMMA_BF16(load_frag(ar + kb, lane), load_frag(br + kb, lane), ai);
        }
        for (int kb = 0; kb < 512; kb += 32)
          ah = WMMA_BF16(load_frag(ar2 + kb, lane), load_frag(br2 + kb, lane), ah);
        for (int r = 0; r < 8; ++r) {
          const int m = m0 + r + ((lane >> 4) << 3);
          gh[2][m][lc] = ai[r];
          gh[3][m][lc] = ah[r];
        }
      }
    }
    __syncthreads();
    for (int idx = threadIdx.x; idx < 64 * 32; idx += 256) {
      const int m = idx >> 5, lc = idx & 31;
      const int col = hc0 + lc;
      const float r_ = 1.f / (1.f + __expf(-(gh[0][m][lc] + bi[col] + bh[col])));
      const float z_ = 1.f / (1.f + __expf(-(gh[1][m][lc] + bi[512 + col] + bh[512 + col])));
      const float n_ = tanhf(gh[2][m][lc] + bi[1024 + col] + r_ * (gh[3][m][lc] + bh[1024 + col]));
      const float hn = (1.f - z_) * n_ + z_ * hloc[idx];
      hloc[idx] = hn;
      const uint16_t hb16 = f2bf(hn);
      hW[m * 512 + col] = hb16;
      y1[(size_t)t * 65536 + m * 1024 + dir * 512 + col] = hb16;
    }
    gbar(bar, gridDim.x);
  }
  for (int idx = threadIdx.x; idx < 64 * 32; idx += 256) {
    const int m = idx >> 5, lc = idx & 31;
    henc[(2 + dir) * 32768 + m * 512 + hc0 + lc] = hloc[idx];
  }
}

// ---------------- decoder attention (batched GEMV; enc_outs bf16 is L2-resident) ----------------
__global__ void __launch_bounds__(256) dec_attn(
    const uint16_t* __restrict__ enc,   // [1024][64][1024]
    const float* __restrict__ q,        // [64][1024]
    float* __restrict__ attw,           // [64][32][1024] (output region)
    uint16_t* __restrict__ attbf,       // [64][1024]
    int step)
{
  const int b = blockIdx.x;
  const int tid = threadIdx.x;
  __shared__ float sq[1024];
  __shared__ float sw[1024];
  __shared__ float red[256];
  for (int i = tid; i < 1024; i += 256) sq[i] = q[b * 1024 + i];
  __syncthreads();
  float sv[4];
  float lmax = -3.402823e38f;
  for (int li = 0; li < 4; ++li) {
    const int l = tid + (li << 8);
    const uint16_t* er = enc + (size_t)l * 65536 + b * 1024;
    float s = 0.f;
    for (int h = 0; h < 1024; h += 8) {
      v4u u = *(const v4u*)(er + h);
      s += bf2f((uint16_t)u[0]) * sq[h]     + bf2f((uint16_t)(u[0] >> 16)) * sq[h + 1]
         + bf2f((uint16_t)u[1]) * sq[h + 2] + bf2f((uint16_t)(u[1] >> 16)) * sq[h + 3]
         + bf2f((uint16_t)u[2]) * sq[h + 4] + bf2f((uint16_t)(u[2] >> 16)) * sq[h + 5]
         + bf2f((uint16_t)u[3]) * sq[h + 6] + bf2f((uint16_t)(u[3] >> 16)) * sq[h + 7];
    }
    sv[li] = s;
    lmax = fmaxf(lmax, s);
  }
  red[tid] = lmax; __syncthreads();
  for (int o = 128; o > 0; o >>= 1) { if (tid < o) red[tid] = fmaxf(red[tid], red[tid + o]); __syncthreads(); }
  const float mx = red[0];
  __syncthreads();
  float lsum = 0.f;
  for (int li = 0; li < 4; ++li) { sv[li] = __expf(sv[li] - mx); lsum += sv[li]; }
  red[tid] = lsum; __syncthreads();
  for (int o = 128; o > 0; o >>= 1) { if (tid < o) red[tid] += red[tid + o]; __syncthreads(); }
  const float inv = 1.f / red[0];
  for (int li = 0; li < 4; ++li) {
    const int l = tid + (li << 8);
    const float w = sv[li] * inv;
    sw[l] = w;
    attw[(size_t)b * 32768 + step * 1024 + l] = w;
  }
  __syncthreads();
  float a0 = 0.f, a1 = 0.f, a2 = 0.f, a3 = 0.f;
  const int h0 = tid << 2;
  for (int l = 0; l < 1024; ++l) {
    const uint16_t* er = enc + (size_t)l * 65536 + b * 1024 + h0;
    v2u u = *(const v2u*)er;
    const float w = sw[l];
    a0 += w * bf2f((uint16_t)u[0]);
    a1 += w * bf2f((uint16_t)(u[0] >> 16));
    a2 += w * bf2f((uint16_t)u[1]);
    a3 += w * bf2f((uint16_t)(u[1] >> 16));
  }
  attbf[b * 1024 + h0]     = f2bf(a0);
  attbf[b * 1024 + h0 + 1] = f2bf(a1);
  attbf[b * 1024 + h0 + 2] = f2bf(a2);
  attbf[b * 1024 + h0 + 3] = f2bf(a3);
}

// ---------------- decoder logits: [h1 | att] @ Wout^T + bout  (split-A WMMA) ----------------
__global__ void __launch_bounds__(256) dec_logits(
    const uint16_t* __restrict__ h1bf,   // [64][512]
    const uint16_t* __restrict__ attbf,  // [64][1024]
    const uint16_t* __restrict__ Woutbf, // [128][1536]
    const float* __restrict__ bout,
    float* __restrict__ out,             // vec_out region [64][32][128]
    int step)
{
  const int lane = threadIdx.x & 31;
  const int wave = (blockIdx.x * blockDim.x + threadIdx.x) >> 5;
  if (wave >= 32) return;                   // 4 m-tiles x 8 n-tiles
  const int m0 = (wave >> 3) << 4;
  const int n0 = (wave & 7) << 4;
  v8f acc = {};
  const uint16_t* br = Woutbf + (size_t)(n0 + (lane & 15)) * 1536;
  const uint16_t* a1 = h1bf + (m0 + (lane & 15)) * 512;
  const uint16_t* a2 = attbf + (m0 + (lane & 15)) * 1024;
  for (int kb = 0; kb < 1536; kb += 32) {
    v16bf a = (kb < 512) ? load_frag(a1 + kb, lane) : load_frag(a2 + (kb - 512), lane);
    v16bf b = load_frag(br + kb, lane);
    acc = WMMA_BF16(a, b, acc);
  }
  const int col = n0 + (lane & 15);
  for (int r = 0; r < 8; ++r) {
    const int row = m0 + r + ((lane >> 4) << 3);
    out[(size_t)row * 4096 + step * 128 + col] = acc[r] + bout[col];
  }
}

// ---------------- argmax over 128 logits (first-max tie-break like jnp.argmax) ----------------
__global__ void dec_argmax(const float* __restrict__ out, int step, int* __restrict__ tok) {
  const int b = blockIdx.x;
  const int d = threadIdx.x;          // 128 threads
  __shared__ float sv[128];
  __shared__ int   si[128];
  sv[d] = out[(size_t)b * 4096 + step * 128 + d];
  si[d] = d;
  __syncthreads();
  for (int o = 64; o > 0; o >>= 1) {
    if (d < o) {
      if (sv[d + o] > sv[d] || (sv[d + o] == sv[d] && si[d + o] < si[d])) {
        sv[d] = sv[d + o]; si[d] = si[d + o];
      }
    }
    __syncthreads();
  }
  if (d == 0) tok[b] = si[0];
}

// ---------------- host orchestration ----------------
extern "C" void kernel_launch(void* const* d_in, const int* in_sizes, int n_in,
                              void* d_out, int out_size, void* d_ws, size_t ws_size,
                              hipStream_t stream) {
  const float* x       = (const float*)d_in[0];
  const float* Wih0    = (const float*)d_in[1];
  const float* Whh0    = (const float*)d_in[2];
  const float* bih0    = (const float*)d_in[3];
  const float* bhh0    = (const float*)d_in[4];
  const float* Wih1    = (const float*)d_in[5];
  const float* Whh1    = (const float*)d_in[6];
  const float* bih1    = (const float*)d_in[7];
  const float* bhh1    = (const float*)d_in[8];
  const float* emb     = (const float*)d_in[9];
  const float* dWih    = (const float*)d_in[10];
  const float* dWhh    = (const float*)d_in[11];
  const float* dbih    = (const float*)d_in[12];
  const float* dbhh    = (const float*)d_in[13];
  const float* Wq      = (const float*)d_in[14];
  const float* bq      = (const float*)d_in[15];
  const float* We2d    = (const float*)d_in[16];
  const float* be2d    = (const float*)d_in[17];
  const float* Wout    = (const float*)d_in[18];
  const float* bout    = (const float*)d_in[19];
  float* out = (float*)d_out;
  float* out_vec  = out;               // [64,32,128]
  float* out_hidf = out + 262144;      // [2,64,512]
  float* out_attw = out + 327680;      // [64,32,1024]

  // workspace carve-out (~290 MB; enc_outs kept in bf16 so it stays L2-resident)
  uint8_t* p = (uint8_t*)d_ws;
  auto alloc = [&](size_t bytes) { uint8_t* r = p; p += (bytes + 255) & ~(size_t)255; return r; };
  uint16_t* Whh0bf = (uint16_t*)alloc(1572864ull * 2);   // [2,1536,512]
  uint16_t* Wih1bf = (uint16_t*)alloc(3145728ull * 2);   // [2,1536,1024]
  uint16_t* Whh1bf = (uint16_t*)alloc(1572864ull * 2);
  uint16_t* dWihbf = (uint16_t*)alloc(524288ull * 2);    // [2,512,512]
  uint16_t* dWhhbf = (uint16_t*)alloc(524288ull * 2);
  uint16_t* Wqbf   = (uint16_t*)alloc(524288ull * 2);    // [1024,512]
  uint16_t* Woutbf = (uint16_t*)alloc(196608ull * 2);    // [128,1536]
  uint16_t* We2dbf = (uint16_t*)alloc(2097152ull * 2);   // [1024,2048]
  uint16_t* embbf  = (uint16_t*)alloc(65536ull * 2);     // [128,512]
  uint16_t* y0bf   = (uint16_t*)alloc(67108864ull * 2);  // [1024,64,1024] 128MB
  uint16_t* y1bf   = (uint16_t*)alloc(67108864ull * 2);  // [1024,64,1024] 128MB (enc_outs)
  uint16_t* hbf    = (uint16_t*)alloc(131072ull * 2);    // [2 parity][2 dir][64][512]
  float*    henc   = (float*)alloc(131072ull * 4);       // [4,64,512] == hflat [64,2048]
  uint16_t* hencbf = (uint16_t*)alloc(131072ull * 2);
  float*    Rbuf   = (float*)alloc(65536ull * 4);        // [64,1024] == hdec [2,64,512]
  float*    hdF    = (float*)alloc(2ull * 65536 * 4);    // [parity][2,64,512] fp32
  uint16_t* hdB    = (uint16_t*)alloc(2ull * 65536 * 2); // bf16 mirror
  uint16_t* xinbf  = (uint16_t*)alloc(32768ull * 2);     // [64,512]
  float*    qf     = (float*)alloc(65536ull * 4);        // [64,1024]
  uint16_t* attbf  = (uint16_t*)alloc(65536ull * 2);     // [64,1024]
  int*      tok    = (int*)alloc(64 * 4);
  int*      bar    = (int*)alloc(8 * 4);
  (void)ws_size; (void)n_in; (void)in_sizes; (void)out_size;

  auto CVT = [&](const float* s, uint16_t* d, int n) {
    cvt_bf<<<(n + 255) / 256, 256, 0, stream>>>(s, d, n);
  };
  // weight conversions (fp32 -> bf16, row-major [N,K] == WMMA B operand)
  CVT(Whh0, Whh0bf, 1572864);
  CVT(Wih1, Wih1bf, 3145728);
  CVT(Whh1, Whh1bf, 1572864);
  CVT(dWih, dWihbf, 524288);
  CVT(dWhh, dWhhbf, 524288);
  CVT(Wq,   Wqbf,   524288);
  CVT(Wout, Woutbf, 196608);
  CVT(We2d, We2dbf, 2097152);
  CVT(emb,  embbf,  65536);

  // ---- encoder layer 0 ----
  zero32<<<256, 256, 0, stream>>>((int*)hbf, 65536);   // both parities = h0 = 0
  zero32<<<1, 64, 0, stream>>>(bar, 8);
  zero32<<<1, 64, 0, stream>>>(tok, 64);
  enc_gru0<<<32, 256, 0, stream>>>(x, Wih0, bih0, bhh0, Whh0bf, hbf, y0bf, henc, bar);

  // ---- encoder layer 1 ----
  zero32<<<256, 256, 0, stream>>>((int*)hbf, 65536);
  zero32<<<1, 64, 0, stream>>>(bar, 8);
  enc_gru1<<<32, 256, 0, stream>>>(y0bf, Wih1bf, bih1, bhh1, Whh1bf, hbf, y1bf, henc, bar);

  // ---- bridge: hdec = (hflat @ We2d^T + be2d) ; both reshapes are pure reinterprets ----
  CVT(henc, hencbf, 131072);
  wmma_gemm_dual<<<32, 256, 0, stream>>>(hencbf, 2048, We2dbf, 2048, 2048,
                                         nullptr, 0, nullptr, 0, 0,
                                         be2d, nullptr, Rbuf, 1024, nullptr, 0,
                                         64, 1024, 0);
  r2hdec<<<256, 256, 0, stream>>>(Rbuf, hdF, hdB, 65536);   // -> parity 0

  // ---- decoder: 32 sequential steps ----
  for (int s = 0; s < 32; ++s) {
    const int rd = s & 1, wr = 1 - rd;
    float*    hFw = hdF + wr * 65536;
    uint16_t* hBw = hdB + wr * 65536;
    const uint16_t* hBr = hdB + rd * 65536;
    dec_gather<<<128, 256, 0, stream>>>(embbf, tok, xinbf);
    // layer 0: tanh(xin@Wih0^T + bih0 + h0@Whh0^T + bhh0)
    wmma_gemm_dual<<<16, 256, 0, stream>>>(xinbf, 512, dWihbf, 512, 512,
                                           hBr, 512, dWhhbf, 512, 512,
                                           dbih, dbhh, hFw, 512, hBw, 512,
                                           64, 512, 1);
    // layer 1
    wmma_gemm_dual<<<16, 256, 0, stream>>>(hBw, 512, dWihbf + 262144, 512, 512,
                                           hBr + 32768, 512, dWhhbf + 262144, 512, 512,
                                           dbih + 512, dbhh + 512,
                                           hFw + 32768, 512, hBw + 32768, 512,
                                           64, 512, 1);
    // q = vec_rnn @ Wq^T + bq
    wmma_gemm_dual<<<32, 256, 0, stream>>>(hBw + 32768, 512, Wqbf, 512, 512,
                                           nullptr, 0, nullptr, 0, 0,
                                           bq, nullptr, qf, 1024, nullptr, 0,
                                           64, 1024, 0);
    dec_attn<<<64, 256, 0, stream>>>(y1bf, qf, out_attw, attbf, s);
    dec_logits<<<4, 256, 0, stream>>>(hBw + 32768, attbf, Woutbf, bout, out_vec, s);
    dec_argmax<<<64, 128, 0, stream>>>(out_vec, s, tok);
  }
  // step 31 writes parity 0 -> final hidden
  copyf<<<256, 256, 0, stream>>>(hdF, out_hidf, 65536);
}
